// HRP_45664092291289
// MI455X (gfx1250) — compile-verified, matrix operand order checked by reading, not run on hardware
//
#include <hip/hip_runtime.h>
#include <cmath>

typedef __attribute__((ext_vector_type(16))) _Float16 v16h;
typedef __attribute__((ext_vector_type(8)))  float    v8f;

#define NZ   100
#define TT   2
#define SS   6
#define HH   64
#define GG   192   // 3*H gates
#define NBC  35
#define LMHH 128

__device__ __forceinline__ float sigmoidf_(float v){ return 1.0f/(1.0f + expf(-v)); }

__device__ __forceinline__ v8f wmma_f16(v16h a, v16h b, v8f c){
  // D = A(16x32 f16) * B(32x16 f16) + C(16x16 f32)
  return __builtin_amdgcn_wmma_f32_16x16x32_f16(false, a, false, b, (short)0, c, false, false);
}

// A fragment from row-major f16 LDS [16][ld]: lane holds row m=lane&15,
// K pattern: e<8 -> k0+kb8+e ; e>=8 -> k0+16+kb8+(e-8), kb8 = 8*(lane>>4)
__device__ __forceinline__ v16h load_a(const _Float16* buf, int ld, int m, int kb8, int k0){
  v16h a;
  #pragma unroll
  for (int i=0;i<8;++i){
    a[i]   = buf[m*ld + k0 + kb8 + i];
    a[i+8] = buf[m*ld + k0 + 16 + kb8 + i];
  }
  return a;
}

// B fragment straight from global row-major f32 W[G][K] (critic path).
__device__ __forceinline__ v16h load_b(const float* __restrict__ W, int K, int g, int kb8, int k0){
  v16h b;
  const float* p = W + (size_t)g*K + k0 + kb8;
  #pragma unroll
  for (int i=0;i<8;++i){
    b[i]   = (_Float16)p[i];
    b[i+8] = (_Float16)p[16+i];
  }
  return b;
}

// B fragment from a pre-swizzled f16 LDS table: tile = 32 lanes x 16 halves (1KB),
// lane's 16 halves contiguous -> two ds_load_b128.
__device__ __forceinline__ v16h load_b_swz(const _Float16* tbl, int tIdx, int lane){
  const _Float16* p = tbl + tIdx*512 + lane*16;
  v16h b;
  #pragma unroll
  for (int i=0;i<16;++i) b[i] = p[i];
  return b;
}

// One-time per-block: convert W (row-major [G][K] f32, G = 16*njt*ngates) into
// pre-swizzled f16 B-fragment tiles. Tile order: ((jt*ngates + gate)*2 + ks).
// Work item w -> (tile, lane, half-group): 8 consecutive f32 -> 8 f16.
__device__ __forceinline__ void stage_swz(int tid, int nthr, _Float16* dst,
    const float* __restrict__ W, int K, int njt, int ngates){
  const int ntiles = njt*ngates*2;
  for (int w = tid; w < ntiles*64; w += nthr){
    const int tIdx = w >> 6;
    const int r    = w & 63;
    const int lane = r >> 1;
    const int hg   = r & 1;
    const int jt   = tIdx/(ngates*2);
    const int rem  = tIdx%(ngates*2);
    const int gate = rem >> 1;
    const int ks   = rem & 1;
    const int g    = (lane&15) + 16*jt + 64*gate;
    const int kb8  = (lane>>4)*8;
    const float* p = W + (size_t)g*K + ks*32 + kb8 + hg*16;
    _Float16* d = dst + tIdx*512 + lane*16 + hg*8;
    #pragma unroll
    for (int i=0;i<8;++i) d[i] = (_Float16)p[i];
  }
}

__device__ __forceinline__ void relu_stage16(int lane, const _Float16* src, _Float16* dst, int n){
  for (int i=lane;i<n;i+=32){
    float v = (float)src[i];
    dst[i] = (_Float16)(v > 0.f ? v : 0.f);
  }
}

// Y = relu(A @ W^T + bias), B fragments from pre-swizzled LDS table.
template<int KDIM, int NDIM>
__device__ __forceinline__ void fc_relu_swz(int lane, const _Float16* A, _Float16* Y,
                                            const _Float16* tbl, const float* bias){
  const int lo = lane & 15, hi = lane >> 4, kb8 = hi*8;
  #pragma unroll
  for (int jt=0; jt<NDIM/16; ++jt){
    const int col = lo + 16*jt;
    const float bv = bias[col];
    v8f c;
    #pragma unroll
    for (int v=0;v<8;++v) c[v]=bv;
    #pragma unroll
    for (int ks=0; ks<KDIM/32; ++ks){
      v16h a = load_a(A, KDIM, lo, kb8, ks*32);
      v16h b = load_b_swz(tbl, jt*(KDIM/32)+ks, lane);
      c = wmma_f16(a, b, c);
    }
    #pragma unroll
    for (int v=0;v<8;++v){
      float y = c[v]; y = y>0.f ? y : 0.f;
      Y[(hi*8+v)*NDIM + col] = (_Float16)y;
    }
  }
}

// Y = relu(A @ W^T + bias), B fragments converted from global f32 (critic).
template<int KDIM, int NDIM>
__device__ __forceinline__ void fc_relu(int lane, const _Float16* A, _Float16* Y,
                                        const float* __restrict__ Wg, const float* bias){
  const int lo = lane & 15, hi = lane >> 4, kb8 = hi*8;
  #pragma unroll
  for (int jt=0; jt<NDIM/16; ++jt){
    const int col = lo + 16*jt;
    const float bv = bias[col];
    v8f c;
    #pragma unroll
    for (int v=0;v<8;++v) c[v]=bv;
    #pragma unroll
    for (int ks=0; ks<KDIM/32; ++ks){
      v16h a = load_a(A, KDIM, lo, kb8, ks*32);
      v16h b = load_b(Wg, KDIM, col, kb8, ks*32);
      c = wmma_f16(a, b, c);
    }
    #pragma unroll
    for (int v=0;v<8;++v){
      float y = c[v]; y = y>0.f ? y : 0.f;
      Y[(hi*8+v)*NDIM + col] = (_Float16)y;
    }
  }
}

// ---------------- Actor: 100 zones x (GRU(6->64,T=2) + MLP) --------------
// 2 waves / block; all recurrent + MLP weights staged once per block into
// pre-swizzled f16 LDS fragment tables (~40KB), shared by both waves & both t.
__global__ __launch_bounds__(64) void actor_kernel(
    const float* __restrict__ x,
    const float* __restrict__ Wih, const float* __restrict__ Whh,
    const float* __restrict__ bih, const float* __restrict__ bhh,
    const float* __restrict__ W1, const float* __restrict__ b1,
    const float* __restrict__ W2, const float* __restrict__ b2,
    const float* __restrict__ W3, const float* __restrict__ b3,
    float* __restrict__ P, int B)
{
  __shared__ _Float16 swzWhh[24*512];           // 4 jt x 3 gates x 2 ks
  __shared__ _Float16 swzW1[8*512], swzW2[8*512]; // 4 jt x 2 ks
  __shared__ float sWih[GG*SS];
  __shared__ float sbih[GG], sbhh[GG], sb1[HH], sb2[HH], sW3[HH];
  __shared__ float sb3;
  __shared__ float sx[2][16][TT][SS];
  __shared__ _Float16 hA[2][16*HH], hB[2][16*HH], hX[2][16*HH];

  const int n = blockIdx.x;
  const int tid = threadIdx.x, wave = tid>>5, lane = tid&31;
  const int lo = lane&15, hi = lane>>4, kb8 = hi*8;

  const float* Whh_n = Whh + (size_t)n*GG*HH;
  const float* W1_n  = W1  + (size_t)n*HH*HH;
  const float* W2_n  = W2  + (size_t)n*HH*HH;

  // ---- one-time block staging ----
  stage_swz(tid, blockDim.x, swzWhh, Whh_n, HH, 4, 3);
  stage_swz(tid, blockDim.x, swzW1,  W1_n,  HH, 4, 1);
  stage_swz(tid, blockDim.x, swzW2,  W2_n,  HH, 4, 1);
  for (int i=tid;i<GG*SS;i+=blockDim.x) sWih[i] = Wih[(size_t)n*GG*SS + i];
  for (int i=tid;i<GG;i+=blockDim.x){ sbih[i]=bih[(size_t)n*GG+i]; sbhh[i]=bhh[(size_t)n*GG+i]; }
  for (int i=tid;i<HH;i+=blockDim.x){ sb1[i]=b1[n*HH+i]; sb2[i]=b2[n*HH+i]; sW3[i]=W3[n*HH+i]; }
  if (tid==0) sb3 = b3[n];
  __syncthreads();

  const int b0 = blockIdx.y*32 + wave*16;
  for (int i=lane;i<16*TT*SS;i+=32){
    int m=i/(TT*SS), r=i%(TT*SS), t=r/SS, s=r%SS;
    sx[wave][m][t][s] = x[(((size_t)(b0+m)*TT + t)*NZ + n)*SS + s];
  }

  _Float16* A_ = hA[wave]; _Float16* B_ = hB[wave]; _Float16* X_ = hX[wave];

  // ---- t=0 GRU cell: h_prev=0 => gh = bhh (pure elementwise) ----
  #pragma unroll
  for (int jt=0;jt<4;++jt){
    const int col = lo + 16*jt;
    float wr[SS],wz[SS],wn[SS];
    #pragma unroll
    for (int s=0;s<SS;++s){ wr[s]=sWih[col*SS+s]; wz[s]=sWih[(64+col)*SS+s]; wn[s]=sWih[(128+col)*SS+s]; }
    const float br=sbih[col], bz=sbih[64+col], bn=sbih[128+col];
    const float hr=sbhh[col], hz=sbhh[64+col], hn=sbhh[128+col];
    #pragma unroll
    for (int v=0;v<8;++v){
      const int m = hi*8+v;
      float gr=br, gz=bz, gn=bn;
      #pragma unroll
      for (int s=0;s<SS;++s){ float xv=sx[wave][m][0][s]; gr+=xv*wr[s]; gz+=xv*wz[s]; gn+=xv*wn[s]; }
      float r = sigmoidf_(gr+hr), z = sigmoidf_(gz+hz);
      float nn = tanhf(gn + r*hn);
      A_[m*HH+col] = (_Float16)((1.f-z)*nn);       // raw h1
    }
  }

  // ---- MLP head, t=0 ----
  relu_stage16(lane, A_, B_, 16*HH);
  fc_relu_swz<64,64>(lane, B_, X_, swzW1, sb1);
  fc_relu_swz<64,64>(lane, X_, B_, swzW2, sb2);
  if (lane < 16){
    float acc = sb3;
    #pragma unroll
    for (int k=0;k<HH;++k) acc += (float)B_[lane*HH+k]*sW3[k];
    P[((size_t)n*B + (b0+lane))*TT + 0] = 3.0f*(sigmoidf_(acc)-0.5f);
  }

  // ---- t=1: gh = h1 @ Whh^T + bhh via WMMA (bias in accumulator), fused cell ----
  #pragma unroll
  for (int jt=0;jt<4;++jt){
    const int col = lo + 16*jt;
    v8f cr,cz,cn;
    { const float a0=sbhh[col], a1=sbhh[64+col], a2=sbhh[128+col];
      #pragma unroll
      for (int v=0;v<8;++v){ cr[v]=a0; cz[v]=a1; cn[v]=a2; } }
    #pragma unroll
    for (int ks=0;ks<2;++ks){
      v16h a = load_a(A_, HH, lo, kb8, ks*32);
      cr = wmma_f16(a, load_b_swz(swzWhh, (jt*3+0)*2+ks, lane), cr);
      cz = wmma_f16(a, load_b_swz(swzWhh, (jt*3+1)*2+ks, lane), cz);
      cn = wmma_f16(a, load_b_swz(swzWhh, (jt*3+2)*2+ks, lane), cn);
    }
    float wr[SS],wz[SS],wn[SS];
    #pragma unroll
    for (int s=0;s<SS;++s){ wr[s]=sWih[col*SS+s]; wz[s]=sWih[(64+col)*SS+s]; wn[s]=sWih[(128+col)*SS+s]; }
    const float br=sbih[col], bz=sbih[64+col], bn=sbih[128+col];
    #pragma unroll
    for (int v=0;v<8;++v){
      const int m = hi*8+v;
      float gr=br, gz=bz, gn=bn;
      #pragma unroll
      for (int s=0;s<SS;++s){ float xv=sx[wave][m][1][s]; gr+=xv*wr[s]; gz+=xv*wz[s]; gn+=xv*wn[s]; }
      float hp = (float)A_[m*HH+col];
      float r = sigmoidf_(gr+cr[v]), z = sigmoidf_(gz+cz[v]);
      float nn = tanhf(gn + r*cn[v]);
      X_[m*HH+col] = (_Float16)((1.f-z)*nn + z*hp); // raw h2
    }
  }

  // ---- MLP head, t=1 ----
  relu_stage16(lane, X_, B_, 16*HH);
  fc_relu_swz<64,64>(lane, B_, A_, swzW1, sb1);
  fc_relu_swz<64,64>(lane, A_, B_, swzW2, sb2);
  if (lane < 16){
    float acc = sb3;
    #pragma unroll
    for (int k=0;k<HH;++k) acc += (float)B_[lane*HH+k]*sW3[k];
    P[((size_t)n*B + (b0+lane))*TT + 1] = 3.0f*(sigmoidf_(acc)-0.5f);
  }
}

// -------- p[:, -1] :  out[b*100+nz] = Pflat[b*200 + 100 + nz] ----------
__global__ void plast_kernel(const float* __restrict__ P, float* __restrict__ out, int B){
  int i = blockIdx.x*blockDim.x + threadIdx.x;
  if (i < B*NZ){
    int b = i/NZ, nz = i%NZ;
    out[i] = P[(size_t)b*(TT*NZ) + NZ + nz];
  }
}

// -------- Critic (zone 99 only) + LocalizedModule 99 ------------------
__global__ __launch_bounds__(64) void critic_kernel(
    const float* __restrict__ x, const float* __restrict__ P,
    const float* __restrict__ scWih, const float* __restrict__ scWhh,
    const float* __restrict__ scbih, const float* __restrict__ scbhh,
    const float* __restrict__ scW1, const float* __restrict__ scb1,
    const float* __restrict__ scW2, const float* __restrict__ scb2,
    const float* __restrict__ scW3, const float* __restrict__ scb3,
    const float* __restrict__ lmW1, const float* __restrict__ lmb1,
    const float* __restrict__ lmW2, const float* __restrict__ lmb2,
    const float* __restrict__ lmW3, const float* __restrict__ lmb3,
    float* __restrict__ out, int B)
{
  __shared__ float sbih[GG], sbhh[GG], sb1[HH], sb2[HH], sW3[HH];
  __shared__ float slb1[LMHH], slb2[LMHH], slW3[LMHH];
  __shared__ float sb3, slb3;
  __shared__ float sxi[2][16][TT][NBC];
  __shared__ _Float16 hA[2][16*HH], hB[2][16*HH], hX[2][16*HH];
  __shared__ _Float16 l1b[2][16*LMHH], l2b[2][16*LMHH];

  const int tid=threadIdx.x, wave=tid>>5, lane=tid&31;
  const int lo=lane&15, hi=lane>>4, kb8=hi*8;

  for (int i=tid;i<GG;i+=blockDim.x){ sbih[i]=scbih[i]; sbhh[i]=scbhh[i]; }
  for (int i=tid;i<HH;i+=blockDim.x){ sb1[i]=scb1[i]; sb2[i]=scb2[i]; sW3[i]=scW3[i]; }
  for (int i=tid;i<LMHH;i+=blockDim.x){ slb1[i]=lmb1[i]; slb2[i]=lmb2[i]; slW3[i]=lmW3[i]; }
  if (tid==0){ sb3=scb3[0]; slb3=lmb3[0]; }
  __syncthreads();

  const int b0 = blockIdx.x*32 + wave*16;

  // xi = [x99, p99, x89, p89, zeros(14), x98, p98] per (m,t); P read as [B][T][N]
  for (int i=lane;i<16*TT*NBC;i+=32){
    int m=i/(TT*NBC), r=i%(TT*NBC), t=r/NBC, s=r%NBC;
    int b=b0+m;
    float v;
    if (s<6)        v = x[(((size_t)b*TT+t)*NZ + 99)*SS + s];
    else if (s==6)  v = P[((size_t)b*TT+t)*NZ + 99];
    else if (s<13)  v = x[(((size_t)b*TT+t)*NZ + 89)*SS + (s-7)];
    else if (s==13) v = P[((size_t)b*TT+t)*NZ + 89];
    else if (s<28)  v = 0.f;
    else if (s<34)  v = x[(((size_t)b*TT+t)*NZ + 98)*SS + (s-28)];
    else            v = P[((size_t)b*TT+t)*NZ + 98];
    sxi[wave][m][t][s] = v;
  }

  _Float16 *A_=hA[wave], *B_=hB[wave], *X_=hX[wave];

  // ---- GRU t=0 (gh = bhh) ----
  for (int jt=0;jt<4;++jt){
    const int col=lo+16*jt;
    const float hr=sbhh[col], hz=sbhh[64+col], hn=sbhh[128+col];
    float g[3][8];
    for (int gi=0;gi<3;++gi){
      const float* wrow = scWih + (size_t)(gi*64+col)*NBC;
      const float bb = sbih[gi*64+col];
      #pragma unroll
      for (int v=0;v<8;++v){
        const int m=hi*8+v;
        float acc=bb;
        for (int s=0;s<NBC;++s) acc += sxi[wave][m][0][s]*wrow[s];
        g[gi][v]=acc;
      }
    }
    #pragma unroll
    for (int v=0;v<8;++v){
      const int m=hi*8+v;
      float r=sigmoidf_(g[0][v]+hr), z=sigmoidf_(g[1][v]+hz);
      float nn=tanhf(g[2][v]+r*hn);
      A_[m*HH+col]=(_Float16)((1.f-z)*nn);     // raw h1
    }
  }

  // ---- GRU t=1: WMMA recurrence + fused cell ----
  for (int jt=0;jt<4;++jt){
    const int col=lo+16*jt;
    v8f cr,cz,cn;
    { const float a0=sbhh[col],a1=sbhh[64+col],a2=sbhh[128+col];
      #pragma unroll
      for (int v=0;v<8;++v){cr[v]=a0;cz[v]=a1;cn[v]=a2;} }
    #pragma unroll
    for (int ks=0;ks<2;++ks){
      v16h a=load_a(A_,HH,lo,kb8,ks*32);
      cr=wmma_f16(a, load_b(scWhh,HH,col,     kb8,ks*32), cr);
      cz=wmma_f16(a, load_b(scWhh,HH,64+col,  kb8,ks*32), cz);
      cn=wmma_f16(a, load_b(scWhh,HH,128+col, kb8,ks*32), cn);
    }
    float g[3][8];
    for (int gi=0;gi<3;++gi){
      const float* wrow = scWih + (size_t)(gi*64+col)*NBC;
      const float bb=sbih[gi*64+col];
      #pragma unroll
      for (int v=0;v<8;++v){
        const int m=hi*8+v; float acc=bb;
        for (int s=0;s<NBC;++s) acc += sxi[wave][m][1][s]*wrow[s];
        g[gi][v]=acc;
      }
    }
    #pragma unroll
    for (int v=0;v<8;++v){
      const int m=hi*8+v;
      const float hp=(float)A_[m*HH+col];
      float r=sigmoidf_(g[0][v]+cr[v]), z=sigmoidf_(g[1][v]+cz[v]);
      float nn=tanhf(g[2][v]+r*cn[v]);
      X_[m*HH+col]=(_Float16)((1.f-z)*nn+z*hp); // raw h2 (== h_last)
    }
  }

  // ---- SubCritic MLP -> q ----
  relu_stage16(lane, X_, B_, 16*HH);
  fc_relu<64,64>(lane, B_, A_, scW1, sb1);
  fc_relu<64,64>(lane, A_, B_, scW2, sb2);
  float qv=0.f;
  if (lane<16){
    float acc=sb3;
    #pragma unroll
    for (int k=0;k<HH;++k) acc += (float)B_[lane*HH+k]*sW3[k];
    qv=acc;
  }

  // ---- LocalizedModule on xi[:, -1] -> f ----
  _Float16* L1=l1b[wave]; _Float16* L2=l2b[wave];
  for (int i=lane;i<16*LMHH;i+=32){
    int m=i/LMHH, j=i%LMHH;
    float acc=slb1[j];
    const float* wrow=lmW1+(size_t)j*NBC;
    for (int s=0;s<NBC;++s) acc+=sxi[wave][m][1][s]*wrow[s];
    L1[i]=(_Float16)(acc>0.f?acc:0.f);
  }
  fc_relu<128,128>(lane, L1, L2, lmW2, slb2);
  if (lane<16){
    float acc=slb3;
    #pragma unroll
    for (int k=0;k<LMHH;++k) acc+=(float)L2[lane*LMHH+k]*slW3[k];
    out[(size_t)B*NZ + (b0+lane)] = acc + qv;   // q_total = f + q
  }
}

extern "C" void kernel_launch(void* const* d_in, const int* in_sizes, int n_in,
                              void* d_out, int out_size, void* d_ws, size_t ws_size,
                              hipStream_t stream) {
  (void)n_in; (void)out_size; (void)ws_size;
  const float* x     = (const float*)d_in[0];
  const float* aWih  = (const float*)d_in[1];
  const float* aWhh  = (const float*)d_in[2];
  const float* abih  = (const float*)d_in[3];
  const float* abhh  = (const float*)d_in[4];
  const float* aW1   = (const float*)d_in[5];
  const float* ab1   = (const float*)d_in[6];
  const float* aW2   = (const float*)d_in[7];
  const float* ab2   = (const float*)d_in[8];
  const float* aW3   = (const float*)d_in[9];
  const float* ab3   = (const float*)d_in[10];
  const float* lmW1  = (const float*)d_in[11];
  const float* lmb1  = (const float*)d_in[12];
  const float* lmW2  = (const float*)d_in[13];
  const float* lmb2  = (const float*)d_in[14];
  const float* lmW3  = (const float*)d_in[15];
  const float* lmb3  = (const float*)d_in[16];
  const float* scWih = (const float*)d_in[17];
  const float* scWhh = (const float*)d_in[18];
  const float* scbih = (const float*)d_in[19];
  const float* scbhh = (const float*)d_in[20];
  const float* scW1  = (const float*)d_in[21];
  const float* scb1  = (const float*)d_in[22];
  const float* scW2  = (const float*)d_in[23];
  const float* scb2  = (const float*)d_in[24];
  const float* scW3  = (const float*)d_in[25];
  const float* scb3  = (const float*)d_in[26];

  const int B = in_sizes[0] / (TT*NZ*SS);     // 2048
  float* P   = (float*)d_ws;                  // [N][B][T] f32 = 1.6 MB scratch
  float* out = (float*)d_out;                 // [B*N] p_last ++ [B] q_total

  actor_kernel<<<dim3(NZ, B/32), 64, 0, stream>>>(
      x, aWih, aWhh, abih, abhh, aW1, ab1, aW2, ab2, aW3, ab3, P, B);

  plast_kernel<<<(B*NZ + 255)/256, 256, 0, stream>>>(P, out, B);

  critic_kernel<<<B/32, 64, 0, stream>>>(
      x, P, scWih, scWhh, scbih, scbhh, scW1, scb1, scW2, scb2, scW3, scb3,
      lmW1, lmb1, lmW2, lmb2, lmW3, lmb3, out, B);
}
